// MultiHeadedSelfAttention_61735859912960
// MI455X (gfx1250) — matrix operand order repaired and checked by reading, not compiled, not run on hardware
//
#include <hip/hip_runtime.h>
#include <hip/hip_bf16.h>
#include <math.h>

typedef __attribute__((ext_vector_type(2)))  float    v2f;
typedef __attribute__((ext_vector_type(8)))  float    v8f;
typedef __attribute__((ext_vector_type(8)))  _Float16 v8h;
typedef __attribute__((ext_vector_type(16))) _Float16 v16h;

#define NPTS 2048
#define CIN  512
#define KNN  40
#define F6   515
#define SEQ  512
#define FPAD 528
#define HD   512
#define NH   8
#define AD   64
#define NEG_SLOPE 0.2f

// ---------------------------------------------------------------------------
// Generic tiled GEMM: C[M,N] (+epilogue) = A[M,K] * B[K,N], f16 WMMA, f32 acc.
// Template-specialized on operand dtype and epilogue to keep the staging loop
// branch-free. Requires K % 32 == 0 (call sites 512, 2048, 64 meet it; 528 does not).
// MODE: 0 plain store, 1 C=acc*alpha, 2 C=max(C, lrelu(acc*es[m]+eb[m])),
//       3 C=acc+eb[n]
// ---------------------------------------------------------------------------
template <int AH, int BH, int MODE>
__global__ __launch_bounds__(128) void gemm_f16_kernel(
    const void* __restrict__ Ap, long long sam, long long sak, long long abat,
    const void* __restrict__ Bp, long long sbk, long long sbn, long long bbat,
    float* __restrict__ Cp, long long scm, long long scn, long long cbat,
    int M, int N, int Kd, float alpha,
    const float* __restrict__ es, const float* __restrict__ eb) {
  __shared__ __align__(16) _Float16 lA[64 * 40];
  __shared__ __align__(16) _Float16 lB[64 * 40];
  const int t  = threadIdx.x;
  const int w  = t >> 5;
  const int l  = t & 31;
  const int hi = (l >> 4) & 1;
  const int lo = l & 15;
  const int m0 = blockIdx.y * 64;
  const int n0 = blockIdx.x * 64;
  const long long aoff = (long long)blockIdx.z * abat;
  const long long boff = (long long)blockIdx.z * bbat;
  const long long coff = (long long)blockIdx.z * cbat;
  const float*    Af = (const float*)Ap;
  const _Float16* Ah = (const _Float16*)Ap;
  const float*    Bf = (const float*)Bp;
  const _Float16* Bh = (const _Float16*)Bp;

  // staging decomposition: thread t covers k-col (t&31), rows (t>>5)+4i
  const int kl = t & 31;
  const int r0 = t >> 5;

  v8f acc0 = {}, acc1 = {}, acc2 = {}, acc3 = {};

  for (int k0 = 0; k0 < Kd; k0 += 32) {
    // ---- stage A (64m x 32k) and B (64n x 32k) into LDS as f16 ----
    const int gk = k0 + kl;
#pragma unroll
    for (int i = 0; i < 16; ++i) {
      const int rl = r0 + i * 4;
      const int gm = m0 + rl;
      float av = 0.f;
      if (gm < M) {
        const long long ia = aoff + (long long)gm * sam + (long long)gk * sak;
        av = AH ? (float)Ah[ia] : Af[ia];
      }
      lA[rl * 40 + kl] = (_Float16)av;
      const int gn = n0 + rl;
      float bv = 0.f;
      if (gn < N) {
        const long long ib = boff + (long long)gk * sbk + (long long)gn * sbn;
        bv = BH ? (float)Bh[ib] : Bf[ib];
      }
      lB[rl * 40 + kl] = (_Float16)bv;
    }
    // prefetch next A k-slab (global_prefetch_b8)
    if (k0 + 32 < Kd) {
      const int gm = m0 + (t & 63);
      if (gm < M) {
        const long long ia = aoff + (long long)gm * sam + (long long)(k0 + 32) * sak;
        __builtin_prefetch((const char*)Ap + (ia << (AH ? 1 : 2)), 0, 0);
      }
    }
    __syncthreads();

    // ---- fragments ----
    // A 16x32 f16 layout: lane(hi,lo): M=lo; halves j=0..7 -> K=8*hi+j,
    // j=8..15 -> K=16+8*hi+(j-8): two contiguous 16B chunks per lane.
    union { v16h v; v8h h[2]; } ua, ub0, ub1, ub2, ub3;
    const _Float16* arow = &lA[(w * 16 + lo) * 40];
    ua.h[0] = *(const v8h*)(arow + 8 * hi);
    ua.h[1] = *(const v8h*)(arow + 16 + 8 * hi);
    // B 32x16 f16 layout: lane(hi,lo): N=lo; halves j=0..7 -> K=8*hi+j,
    // j=8..15 -> K=16+8*hi+(j-8): two 16B chunks per lane. Preload all 4 subtiles, then issue
    // the 4 WMMAs back-to-back so they pipeline.
    const _Float16* brow = &lB[lo * 40 + 8 * hi];
    ub0.h[0] = *(const v8h*)(brow);             ub0.h[1] = *(const v8h*)(brow + 16);
    ub1.h[0] = *(const v8h*)(brow + 16 * 40);   ub1.h[1] = *(const v8h*)(brow + 16 * 40 + 16);
    ub2.h[0] = *(const v8h*)(brow + 32 * 40);   ub2.h[1] = *(const v8h*)(brow + 32 * 40 + 16);
    ub3.h[0] = *(const v8h*)(brow + 48 * 40);   ub3.h[1] = *(const v8h*)(brow + 48 * 40 + 16);

    acc0 = __builtin_amdgcn_wmma_f32_16x16x32_f16(false, ua.v, false, ub0.v,
                                                  (short)0, acc0, false, false);
    acc1 = __builtin_amdgcn_wmma_f32_16x16x32_f16(false, ua.v, false, ub1.v,
                                                  (short)0, acc1, false, false);
    acc2 = __builtin_amdgcn_wmma_f32_16x16x32_f16(false, ua.v, false, ub2.v,
                                                  (short)0, acc2, false, false);
    acc3 = __builtin_amdgcn_wmma_f32_16x16x32_f16(false, ua.v, false, ub3.v,
                                                  (short)0, acc3, false, false);
    __syncthreads();
  }

  // ---- epilogue: C/D 16x16 layout: element r at lane(hi,lo) -> M=r+8*hi, N=lo
#pragma unroll
  for (int ns = 0; ns < 4; ++ns) {
    v8f a = (ns == 0) ? acc0 : (ns == 1) ? acc1 : (ns == 2) ? acc2 : acc3;
#pragma unroll
    for (int r = 0; r < 8; ++r) {
      const int gm = m0 + w * 16 + r + 8 * hi;
      const int gn = n0 + ns * 16 + lo;
      if (gm < M && gn < N) {
        float v = a[r];
        float* cp = Cp + coff + (long long)gm * scm + (long long)gn * scn;
        if (MODE == 1) {
          v *= alpha;
        } else if (MODE == 2) {
          v = v * es[gm] + eb[gm];
          v = (v < 0.f) ? v * NEG_SLOPE : v;
          const float old = *cp;
          v = (old > v) ? old : v;
        } else if (MODE == 3) {
          v += eb[gn];
        }
        *cp = v;
      }
    }
  }
}

// ---------------------------------------------------------------------------
// KNN gram in full fp32 WMMA: D[n,m] = 2*x^T x - xx[n] - xx[m]
// f32 A 16x4: lane(hi,lo): M=lo, a[v] -> K=v+2*hi. B mirrored, N=lo.
// ---------------------------------------------------------------------------
__global__ __launch_bounds__(128) void gram_kernel(const float* __restrict__ x,
                                                   const float* __restrict__ xx,
                                                   float* __restrict__ D) {
  const int t  = threadIdx.x;
  const int w  = t >> 5;
  const int l  = t & 31;
  const int hi = (l >> 4) & 1;
  const int lo = l & 15;
  const int row0 = blockIdx.y * 16;
  const int col0 = blockIdx.x * 64 + w * 16;
  v8f acc = {};
  for (int c = 0; c < CIN; c += 4) {
    const int ka = c + 2 * hi;
    v2f a, b;
    a.x = x[(long long)(ka + 0) * NPTS + row0 + lo];
    a.y = x[(long long)(ka + 1) * NPTS + row0 + lo];
    b.x = x[(long long)(ka + 0) * NPTS + col0 + lo];
    b.y = x[(long long)(ka + 1) * NPTS + col0 + lo];
    acc = __builtin_amdgcn_wmma_f32_16x16x4_f32(false, a, false, b, (short)0,
                                                acc, false, false);
  }
#pragma unroll
  for (int r = 0; r < 8; ++r) {
    const int gr = row0 + r + 8 * hi;
    const int gc = col0 + lo;
    D[(long long)gr * NPTS + gc] = 2.f * acc[r] - xx[gr] - xx[gc];
  }
}

// ---------------------------------------------------------------------------
// xx[n] = sum_c x[c,n]^2
// ---------------------------------------------------------------------------
__global__ void sqnorm_kernel(const float* __restrict__ x, float* __restrict__ xx) {
  const int n = blockIdx.x * blockDim.x + threadIdx.x;
  if (n < NPTS) {
    float s = 0.f;
    for (int c = 0; c < CIN; ++c) {
      const float v = x[(long long)c * NPTS + n];
      s += v * v;
    }
    xx[n] = s;
  }
}

// ---------------------------------------------------------------------------
// Top-40 per row (wave per row, iterative argmax, lowest-index tie break)
// ---------------------------------------------------------------------------
__global__ void topk_kernel(const float* __restrict__ D, int* __restrict__ idx) {
  __shared__ float rows[4][NPTS];
  const int w = threadIdx.x >> 5;
  const int l = threadIdx.x & 31;
  const int row = blockIdx.x * 4 + w;
  float* r = rows[w];
  for (int j = l; j < NPTS; j += 32) r[j] = D[(long long)row * NPTS + j];
  __syncthreads();
  for (int it = 0; it < KNN; ++it) {
    float bv = -3.4e38f;
    int bi = NPTS;
    for (int j = l; j < NPTS; j += 32) {
      const float v = r[j];
      if (v > bv) { bv = v; bi = j; }
    }
    for (int off = 16; off; off >>= 1) {
      const float ov = __shfl_xor(bv, off, 32);
      const int   oi = __shfl_xor(bi, off, 32);
      if (ov > bv || (ov == bv && oi < bi)) { bv = ov; bi = oi; }
    }
    if (l == 0) {
      idx[row * KNN + it] = bi;
      r[bi] = -3.4e38f;
    }
    __syncthreads();
  }
}

// ---------------------------------------------------------------------------
// BN affine coefficients
// ---------------------------------------------------------------------------
__global__ void bncoef_kernel(const float* g5, const float* b5, const float* m5,
                              const float* v5, const float* g6, const float* b6,
                              const float* m6, const float* v6, float* s5o,
                              float* b5o, float* s6o, float* b6o) {
  const int i = blockIdx.x * blockDim.x + threadIdx.x;
  if (i < 512) {
    const float s = g5[i] / sqrtf(v5[i] + 1e-5f);
    s5o[i] = s;
    b5o[i] = b5[i] - m5[i] * s;
  }
  if (i < F6) {
    const float s = g6[i] / sqrtf(v6[i] + 1e-5f);
    s6o[i] = s;
    b6o[i] = b6[i] - m6[i] * s;
  }
}

__global__ void w5diff_kernel(const float* __restrict__ w5, float* __restrict__ d) {
  const int t = blockIdx.x * blockDim.x + threadIdx.x; // 512*512
  const int s = t >> 9, c = t & 511;
  d[t] = w5[s * 1024 + 512 + c] - w5[s * 1024 + c];
}

__global__ void padw_kernel(const float* __restrict__ W, float* __restrict__ Wp) {
  const int t = blockIdx.x * blockDim.x + threadIdx.x; // 512*528
  const int o = t / FPAD, f = t - o * FPAD;
  Wp[t] = (f < F6) ? W[o * F6 + f] : 0.f;
}

__global__ void initf_kernel(float* __restrict__ fp) {
  const int t = blockIdx.x * blockDim.x + threadIdx.x; // 512*528
  const int f = t % FPAD;
  fp[t] = (f < F6) ? -3.4e38f : 0.f;
}

// ---------------------------------------------------------------------------
// hbk[n,s] = lrelu(bn5_s(P[s,idx[n,kk]] + Q[s,n])), f16, for one neighbor kk
// ---------------------------------------------------------------------------
__global__ void build_hbk_kernel(const float* __restrict__ Pt,
                                 const float* __restrict__ Qt,
                                 const int* __restrict__ idx,
                                 const float* __restrict__ s5,
                                 const float* __restrict__ b5,
                                 _Float16* __restrict__ hbk, int kk) {
  const int t = blockIdx.x * blockDim.x + threadIdx.x; // 2048*512
  const int n = t >> 9, s = t & 511;
  const int j = idx[n * KNN + kk];
  float v = Pt[(long long)j * 512 + s] + Qt[(long long)n * 512 + s];
  v = v * s5[s] + b5[s];
  v = (v < 0.f) ? v * NEG_SLOPE : v;
  hbk[t] = (_Float16)v;
}

// ---------------------------------------------------------------------------
// softmax over 512-long rows (block per row)
// ---------------------------------------------------------------------------
__global__ void softmax_kernel(float* __restrict__ sc) {
  __shared__ float red[128];
  const int t = threadIdx.x;
  float* p = sc + (long long)blockIdx.x * 512;
  float m = -3.4e38f;
  for (int j = t; j < 512; j += 128) m = fmaxf(m, p[j]);
  red[t] = m;
  __syncthreads();
  for (int o = 64; o; o >>= 1) {
    if (t < o) red[t] = fmaxf(red[t], red[t + o]);
    __syncthreads();
  }
  m = red[0];
  __syncthreads();
  float s = 0.f;
  for (int j = t; j < 512; j += 128) {
    const float e = expf(p[j] - m);
    p[j] = e;
    s += e;
  }
  red[t] = s;
  __syncthreads();
  for (int o = 64; o; o >>= 1) {
    if (t < o) red[t] += red[t + o];
    __syncthreads();
  }
  const float inv = 1.f / red[0];
  for (int j = t; j < 512; j += 128) p[j] *= inv;
}

// ---------------------------------------------------------------------------
extern "C" void kernel_launch(void* const* d_in, const int* in_sizes, int n_in,
                              void* d_out, int out_size, void* d_ws, size_t ws_size,
                              hipStream_t stream) {
  const float* x    = (const float*)d_in[0];
  const float* y    = (const float*)d_in[1];
  const float* w5   = (const float*)d_in[2];
  const float* g5   = (const float*)d_in[3];
  const float* b5   = (const float*)d_in[4];
  const float* m5   = (const float*)d_in[5];
  const float* v5   = (const float*)d_in[6];
  const float* w6   = (const float*)d_in[7];
  const float* g6   = (const float*)d_in[8];
  const float* b6   = (const float*)d_in[9];
  const float* m6   = (const float*)d_in[10];
  const float* v6   = (const float*)d_in[11];
  const float* Wq   = (const float*)d_in[12];
  const float* Wk   = (const float*)d_in[13];
  const float* Wv   = (const float*)d_in[14];
  const float* Wout = (const float*)d_in[15];
  const float* bout = (const float*)d_in[16];
  float* out = (float*)d_out;

  char* p = (char*)d_ws;
  auto alloc = [&](size_t bytes) {
    char* r = p;
    p += (bytes + 255) & ~(size_t)255;
    return (void*)r;
  };
  float*    xx   = (float*)alloc(NPTS * 4);
  float*    D    = (float*)alloc((size_t)NPTS * NPTS * 4);
  int*      idxb = (int*)alloc((size_t)NPTS * KNN * 4);
  float*    w5d  = (float*)alloc(512 * 512 * 4);
  float*    Pt   = (float*)alloc((size_t)NPTS * 512 * 4);
  float*    Qt   = (float*)alloc((size_t)NPTS * 512 * 4);
  _Float16* hbk  = (_Float16*)alloc((size_t)NPTS * 512 * 2);
  float*    fx   = (float*)alloc(SEQ * FPAD * 4);
  float*    fy   = (float*)alloc(SEQ * FPAD * 4);
  float*    wqp  = (float*)alloc(HD * FPAD * 4);
  float*    wkp  = (float*)alloc(HD * FPAD * 4);
  float*    wvp  = (float*)alloc(HD * FPAD * 4);
  float*    s5c  = (float*)alloc(512 * 4);
  float*    b5c  = (float*)alloc(512 * 4);
  float*    s6c  = (float*)alloc(F6 * 4);
  float*    b6c  = (float*)alloc(F6 * 4);
  float*    qb   = (float*)alloc(SEQ * HD * 4);
  float*    kb   = (float*)alloc(SEQ * HD * 4);
  float*    vb   = (float*)alloc(SEQ * HD * 4);
  float*    sc   = (float*)alloc((size_t)NH * SEQ * SEQ * 4);
  float*    mg   = (float*)alloc(SEQ * HD * 4);

  // mode-dispatching GEMM launcher (AH, BH, MODE compile-time specialized)
  auto gemm = [&](int ah, int bh, int mode,
                  const void* A, long long sam, long long sak, long long ab,
                  const void* B, long long sbk, long long sbn, long long bb,
                  float* C, long long scm, long long scn, long long cb,
                  int M, int N, int Kd, int batch, float alpha,
                  const float* es, const float* eb) {
    dim3 g((N + 63) / 64, (M + 63) / 64, batch);
    dim3 b(128);
    if (ah == 0 && bh == 1 && mode == 2) {
      hipLaunchKernelGGL((gemm_f16_kernel<0, 1, 2>), g, b, 0, stream,
                         A, sam, sak, ab, B, sbk, sbn, bb, C, scm, scn, cb,
                         M, N, Kd, alpha, es, eb);
    } else if (ah == 0 && bh == 0 && mode == 1) {
      hipLaunchKernelGGL((gemm_f16_kernel<0, 0, 1>), g, b, 0, stream,
                         A, sam, sak, ab, B, sbk, sbn, bb, C, scm, scn, cb,
                         M, N, Kd, alpha, es, eb);
    } else if (ah == 0 && bh == 0 && mode == 3) {
      hipLaunchKernelGGL((gemm_f16_kernel<0, 0, 3>), g, b, 0, stream,
                         A, sam, sak, ab, B, sbk, sbn, bb, C, scm, scn, cb,
                         M, N, Kd, alpha, es, eb);
    } else {
      hipLaunchKernelGGL((gemm_f16_kernel<0, 0, 0>), g, b, 0, stream,
                         A, sam, sak, ab, B, sbk, sbn, bb, C, scm, scn, cb,
                         M, N, Kd, alpha, es, eb);
    }
  };

  // ---- shared prep ----
  hipLaunchKernelGGL(bncoef_kernel, dim3(3), dim3(256), 0, stream,
                     g5, b5, m5, v5, g6, b6, m6, v6, s5c, b5c, s6c, b6c);
  hipLaunchKernelGGL(w5diff_kernel, dim3(1024), dim3(256), 0, stream, w5, w5d);
  hipLaunchKernelGGL(padw_kernel, dim3(1056), dim3(256), 0, stream, Wq, wqp);
  hipLaunchKernelGGL(padw_kernel, dim3(1056), dim3(256), 0, stream, Wk, wkp);
  hipLaunchKernelGGL(padw_kernel, dim3(1056), dim3(256), 0, stream, Wv, wvp);

  // ---- per-branch pipeline ----
  auto branch = [&](const float* xin, float* fpad) {
    hipLaunchKernelGGL(sqnorm_kernel, dim3(8), dim3(256), 0, stream, xin, xx);
    hipLaunchKernelGGL(gram_kernel, dim3(32, 128), dim3(128), 0, stream, xin, xx, D);
    hipLaunchKernelGGL(topk_kernel, dim3(512), dim3(128), 0, stream, D, idxb);
    // Pt[n,s] = sum_c x[c,n]*W5a[s,c]   (M=2048,N=512,K=512)
    gemm(0, 0, 0, xin, 1, NPTS, 0, w5, 1, 1024, 0, Pt, 512, 1, 0,
         NPTS, 512, 512, 1, 0.f, nullptr, nullptr);
    // Qt[n,s] = sum_c x[c,n]*(W5b-W5a)[s,c]
    gemm(0, 0, 0, xin, 1, NPTS, 0, w5d, 1, 512, 0, Qt, 512, 1, 0,
         NPTS, 512, 512, 1, 0.f, nullptr, nullptr);
    hipLaunchKernelGGL(initf_kernel, dim3(1056), dim3(256), 0, stream, fpad);
    for (int kk = 0; kk < KNN; ++kk) {
      hipLaunchKernelGGL(build_hbk_kernel, dim3(4096), dim3(256), 0, stream,
                         Pt, Qt, idxb, s5c, b5c, hbk, kk);
      // f[s,o'] = max_k lrelu(bn6(sum_n W6[o',n]*hbk[n,s]))
      gemm(0, 1, 2, w6, 2048, 1, 0, hbk, 512, 1, 0, fpad, 1, FPAD, 0,
           F6, 512, NPTS, 1, 0.f, s6c, b6c);
    }
  };
  branch(x, fx);
  branch(y, fy);

  // ---- heads: q = fx @ Wq^T, k/v from fy (K=528, zero-padded) ----
  gemm(0, 0, 0, fx, FPAD, 1, 0, wqp, 1, FPAD, 0, qb, HD, 1, 0,
       SEQ, HD, FPAD, 1, 0.f, nullptr, nullptr);
  gemm(0, 0, 0, fy, FPAD, 1, 0, wkp, 1, FPAD, 0, kb, HD, 1, 0,
       SEQ, HD, FPAD, 1, 0.f, nullptr, nullptr);
  gemm(0, 0, 0, fy, FPAD, 1, 0, wvp, 1, FPAD, 0, vb, HD, 1, 0,
       SEQ, HD, FPAD, 1, 0.f, nullptr, nullptr);

  // ---- scores = (Q K^T) * scale, batched over 8 heads ----
  const float scale = 0.044194173824159216f; // 512^-0.5
  gemm(0, 0, 1, qb, HD, 1, AD, kb, 1, HD, AD, sc, SEQ, 1, (long long)SEQ * SEQ,
       SEQ, SEQ, AD, NH, scale, nullptr, nullptr);
  hipLaunchKernelGGL(softmax_kernel, dim3(NH * SEQ), dim3(128), 0, stream, sc);

  // ---- weighted = softmax(scores) @ V -> merged[s, h*64+d] ----
  gemm(0, 0, 0, sc, SEQ, 1, (long long)SEQ * SEQ, vb, HD, 1, AD,
       mg, HD, 1, AD, SEQ, AD, SEQ, NH, 0.f, nullptr, nullptr);

  // ---- out[s,o] = merged @ Wout^T + bout ----
  gemm(0, 0, 3, mg, HD, 1, 0, Wout, 1, HD, 0, out, F6, 1, 0,
       SEQ, F6, HD, 1, 0.f, nullptr, bout);
}
